// GCNPolicy_26542897889601
// MI455X (gfx1250) — matrix operand order repaired and checked by reading, not compiled, not run on hardware
//
#include <hip/hip_runtime.h>
#include <hip/hip_bf16.h>

// ---------------------------------------------------------------------------
// GCN policy network for MI455X (gfx1250, wave32, WMMA).
// GEMMs: f16 WMMA with f32 accumulation (v_wmma_f32_16x16x32_f16),
// B staged block-wide in LDS (conflict-free padded layout), 32x64 wave tiles.
// ---------------------------------------------------------------------------

typedef __attribute__((ext_vector_type(16))) _Float16 v16h;
typedef __attribute__((ext_vector_type(8)))  _Float16 v8h;
typedef __attribute__((ext_vector_type(8)))  float    v8f;

// ---------------- degree / normalization ----------------
__global__ void k_deg_init(float* deg, int n) {
    int i = blockIdx.x * blockDim.x + threadIdx.x;
    if (i < n) deg[i] = 1.0f;                      // self-loop contribution
}
__global__ void k_deg_edges(float* deg, const int* __restrict__ dst, int e) {
    int i = blockIdx.x * blockDim.x + threadIdx.x;
    if (i < e) atomicAdd(&deg[dst[i]], 1.0f);
}
__global__ void k_deg_fin(float* dinv, int n) {
    int i = blockIdx.x * blockDim.x + threadIdx.x;
    if (i < n) dinv[i] = rsqrtf(fmaxf(dinv[i], 1.0f));
}

// ---------------- f32 -> f16 conversions ----------------
__global__ void k_conv_pad(const float* __restrict__ in, _Float16* __restrict__ out,
                           int rows, int rowsPad, int K) {
    int idx = blockIdx.x * blockDim.x + threadIdx.x;
    if (idx >= rowsPad * K) return;
    int r = idx / K;
    out[idx] = (r < rows) ? (_Float16)in[idx] : (_Float16)0.0f;
}
// Transposed weight: W is [K x N] row-major f32 -> WT [N x K] f16.
__global__ void k_conv_wt(const float* __restrict__ W, _Float16* __restrict__ WT,
                          int K, int N) {
    int idx = blockIdx.x * blockDim.x + threadIdx.x;
    if (idx >= N * K) return;
    int n = idx / K;
    int k = idx - n * K;
    WT[idx] = (_Float16)W[(size_t)k * N + n];
}

// ---------------- WMMA GEMM ----------------
#define KCHUNK 256
#define LDS_STRIDE (KCHUNK + 8)   // +8 halves (16B) pad -> lanes hit banks 0,4,..,60

__device__ __forceinline__ v16h load_a_frag(const _Float16* p) {
    // A 16x32 fragment, this code's k order (not the ISA table order): elems 0..15 -> K = 16*half + i (contiguous)
    v16h r;
    ((v8h*)&r)[0] = *(const v8h*)(p);
    ((v8h*)&r)[1] = *(const v8h*)(p + 8);
    return r;
}
__device__ __forceinline__ v16h load_b_lds(const _Float16* p) {
    // B 32x16 fragment, same k order as A (not the ISA table order): elems 0..15 -> K = 16*half + i (contiguous, 2x ds_load_b128)
    v16h r;
    ((v8h*)&r)[0] = *(const v8h*)(p);
    ((v8h*)&r)[1] = *(const v8h*)(p + 8);
    return r;
}

// C[Mpad x Nn] (f32) = A16[Mpad x K] (f16, zero-padded rows) * BT16[Nn x K]^T
// Block: 256 threads = 8 waves. Each wave: 32(M) x 64(N) tile. Block: 256(M) x 64(N).
// B strip (64 x K) staged into LDS in K-chunks, shared by all waves.
__global__ void __launch_bounds__(256)
k_wmma_gemm(const _Float16* __restrict__ A16,
            const _Float16* __restrict__ BT16,
            float* __restrict__ C,
            int Mpad, int Nn, int K) {
    __shared__ _Float16 ldsB[64 * LDS_STRIDE];

    const int tid  = threadIdx.x;
    const int lane = tid & 31;
    const int wave = tid >> 5;
    const int half = lane >> 4;
    const int r16  = lane & 15;

    const int m0 = (blockIdx.x * 8 + wave) * 32;   // wave-uniform
    const int n0 = blockIdx.y * 64;
    const bool active = (m0 < Mpad);

    v8f c00 = {}, c01 = {}, c02 = {}, c03 = {};
    v8f c10 = {}, c11 = {}, c12 = {}, c13 = {};

    const _Float16* pa0 = A16 + (size_t)(m0 + r16)      * K + half * 16;
    const _Float16* pa1 = A16 + (size_t)(m0 + 16 + r16) * K + half * 16;
    // per-lane LDS base for B fragments (row = sub*16 + r16, col = 16*half + k)
    const _Float16* lb = ldsB + r16 * LDS_STRIDE + half * 16;

    for (int kc = 0; kc < K; kc += KCHUNK) {
        const int kcLen = (K - kc < KCHUNK) ? (K - kc) : KCHUNK;
        const int c8n = kcLen >> 3;                 // v8h elements per row
        // cooperative stage: 64 rows x kcLen halves, 16B vectors, coalesced
        for (int idx = tid; idx < 64 * c8n; idx += 256) {
            int row = idx / c8n;
            int c8  = idx - row * c8n;
            *(v8h*)(ldsB + row * LDS_STRIDE + c8 * 8) =
                *(const v8h*)(BT16 + (size_t)(n0 + row) * K + kc + c8 * 8);
        }
        __syncthreads();

        if (active) {
            for (int k0 = 0; k0 < kcLen; k0 += 32) {
                v16h a0 = load_a_frag(pa0 + kc + k0);
                v16h a1 = load_a_frag(pa1 + kc + k0);
                v16h b0 = load_b_lds(lb + 0  * 16 * LDS_STRIDE + k0);
                v16h b1 = load_b_lds(lb + 1  * 16 * LDS_STRIDE + k0);
                v16h b2 = load_b_lds(lb + 2  * 16 * LDS_STRIDE + k0);
                v16h b3 = load_b_lds(lb + 3  * 16 * LDS_STRIDE + k0);
                c00 = __builtin_amdgcn_wmma_f32_16x16x32_f16(false, a0, false, b0, (short)0, c00, false, false);
                c01 = __builtin_amdgcn_wmma_f32_16x16x32_f16(false, a0, false, b1, (short)0, c01, false, false);
                c02 = __builtin_amdgcn_wmma_f32_16x16x32_f16(false, a0, false, b2, (short)0, c02, false, false);
                c03 = __builtin_amdgcn_wmma_f32_16x16x32_f16(false, a0, false, b3, (short)0, c03, false, false);
                c10 = __builtin_amdgcn_wmma_f32_16x16x32_f16(false, a1, false, b0, (short)0, c10, false, false);
                c11 = __builtin_amdgcn_wmma_f32_16x16x32_f16(false, a1, false, b1, (short)0, c11, false, false);
                c12 = __builtin_amdgcn_wmma_f32_16x16x32_f16(false, a1, false, b2, (short)0, c12, false, false);
                c13 = __builtin_amdgcn_wmma_f32_16x16x32_f16(false, a1, false, b3, (short)0, c13, false, false);
            }
        }
        __syncthreads();
    }

    if (!active) return;
    // D layout: element j -> row +(j + 8*half), col n0 + r16 (+16 per sub-tile)
    float* pc0 = C + (size_t)(m0 + 8 * half)      * Nn + n0 + r16;
    float* pc1 = C + (size_t)(m0 + 16 + 8 * half) * Nn + n0 + r16;
    #pragma unroll
    for (int j = 0; j < 8; ++j) {
        pc0[(size_t)j * Nn +  0] = c00[j];
        pc0[(size_t)j * Nn + 16] = c01[j];
        pc0[(size_t)j * Nn + 32] = c02[j];
        pc0[(size_t)j * Nn + 48] = c03[j];
        pc1[(size_t)j * Nn +  0] = c10[j];
        pc1[(size_t)j * Nn + 16] = c11[j];
        pc1[(size_t)j * Nn + 32] = c12[j];
        pc1[(size_t)j * Nn + 48] = c13[j];
    }
}

// ---------------- GCN aggregation ----------------
__global__ void k_self_init(const float* __restrict__ h, float* __restrict__ out,
                            const float* __restrict__ dinv, int n, int H) {
    int idx = blockIdx.x * blockDim.x + threadIdx.x;
    int H4 = H >> 2;
    if (idx >= n * H4) return;
    int i = idx / H4;
    float s = dinv[i] * dinv[i];
    float4 v = ((const float4*)h)[idx];
    float4 o; o.x = v.x * s; o.y = v.y * s; o.z = v.z * s; o.w = v.w * s;
    ((float4*)out)[idx] = o;
}
__global__ void k_edge_agg(const float* __restrict__ h, float* __restrict__ out,
                           const int* __restrict__ src, const int* __restrict__ dst,
                           const float* __restrict__ dinv, int e, int H) {
    int idx = blockIdx.x * blockDim.x + threadIdx.x;
    int H4 = H >> 2;
    int ei = idx / H4;
    int r  = idx - ei * H4;
    if (ei >= e) return;
    int s = src[ei], d = dst[ei];
    float sc = dinv[s] * dinv[d];
    float4 v = ((const float4*)(h + (size_t)s * H))[r];
    float* o = out + (size_t)d * H + r * 4;
    atomicAdd(o + 0, v.x * sc);
    atomicAdd(o + 1, v.y * sc);
    atomicAdd(o + 2, v.z * sc);
    atomicAdd(o + 3, v.w * sc);
}
__global__ void k_bias_act(float* h, const float* __restrict__ b, int n, int H, int relu) {
    int idx = blockIdx.x * blockDim.x + threadIdx.x;
    if (idx >= n * H) return;
    int f = idx % H;
    float v = h[idx] + b[f];
    if (relu) v = fmaxf(v, 0.0f);
    h[idx] = v;
}

// ---------------- head + pooling ----------------
__global__ void k_head(const float* __restrict__ h, const float* __restrict__ Wl,
                       const float* __restrict__ bl, float* __restrict__ o,
                       int n, int H, int A) {
    int idx = blockIdx.x * blockDim.x + threadIdx.x;
    if (idx >= n * A) return;
    int i = idx / A;
    int a = idx - i * A;
    const float* hr = h + (size_t)i * H;
    float acc = bl[a];
    for (int k = 0; k < H; ++k) acc = fmaf(hr[k], Wl[k * A + a], acc);
    o[idx] = tanhf(acc);
}
__global__ void k_pool_zero(float* out, float* cnt, int no, int ng) {
    int i = blockIdx.x * blockDim.x + threadIdx.x;
    if (i < no) out[i] = 0.0f;
    if (i < ng) cnt[i] = 0.0f;
}
__global__ void k_pool(const float* __restrict__ h3, const int* __restrict__ batch,
                       float* out, float* cnt, int n, int A) {
    int i = blockIdx.x * blockDim.x + threadIdx.x;
    if (i >= n) return;
    int g = batch[i];
    atomicAdd(&cnt[g], 1.0f);
    for (int a = 0; a < A; ++a) atomicAdd(&out[g * A + a], h3[(size_t)i * A + a]);
}
__global__ void k_pool_fin(float* out, const float* __restrict__ cnt, int no, int A) {
    int i = blockIdx.x * blockDim.x + threadIdx.x;
    if (i >= no) return;
    out[i] /= fmaxf(cnt[i / A], 1.0f);
}

// ---------------------------------------------------------------------------
extern "C" void kernel_launch(void* const* d_in, const int* in_sizes, int n_in,
                              void* d_out, int out_size, void* d_ws, size_t ws_size,
                              hipStream_t stream) {
    const float* x    = (const float*)d_in[0];
    const int*   eidx = (const int*)  d_in[1];
    const int*   batch= (const int*)  d_in[2];
    const float* W1   = (const float*)d_in[3];
    const float* b1   = (const float*)d_in[4];
    const float* W2   = (const float*)d_in[5];
    const float* b2   = (const float*)d_in[6];
    const float* Wl   = (const float*)d_in[7];
    const float* bl   = (const float*)d_in[8];
    float* out = (float*)d_out;

    const int N    = in_sizes[2];          // 10000 nodes
    const int E    = in_sizes[1] / 2;      // 80000 edges
    const int D    = in_sizes[0] / N;      // 128
    const int H    = in_sizes[4];          // 1024
    const int A    = in_sizes[8];          // 8
    const int NG   = out_size / A;         // 64
    const int Mpad = (N + 31) & ~31;       // 10016 (multiple of 32 for wave tiles)

    const int* src = eidx;
    const int* dst = eidx + E;

    // workspace carve-up (256B aligned)
    char* base = (char*)d_ws;
    size_t off = 0;
    auto carve = [&](size_t bytes) -> char* {
        char* p = base + off;
        off = (off + bytes + 255) & ~(size_t)255;
        return p;
    };
    float*    dinv = (float*)   carve((size_t)N * 4);
    float*    hA   = (float*)   carve((size_t)Mpad * H * 4);   // GEMM output
    float*    hB   = (float*)   carve((size_t)Mpad * H * 4);   // aggregation output
    _Float16* A16  = (_Float16*)carve((size_t)Mpad * H * 2);   // f16 GEMM input (padded)
    _Float16* WT16 = (_Float16*)carve((size_t)H * H * 2);      // f16 transposed weights
    float*    h3   = (float*)   carve((size_t)N * A * 4);
    float*    cnt  = (float*)   carve((size_t)NG * 4);
    (void)ws_size; (void)n_in;

    const int TB = 256;
    auto gr = [](long long work, int tb) { return (unsigned)((work + tb - 1) / tb); };

    // 1) degree -> dinv
    k_deg_init <<<gr(N, TB), TB, 0, stream>>>(dinv, N);
    k_deg_edges<<<gr(E, TB), TB, 0, stream>>>(dinv, dst, E);
    k_deg_fin  <<<gr(N, TB), TB, 0, stream>>>(dinv, N);

    // 2) layer 1 GEMM: hA = x @ W1  (f16 WMMA, f32 accum)
    k_conv_pad<<<gr((long long)Mpad * D, TB), TB, 0, stream>>>(x, A16, N, Mpad, D);
    k_conv_wt <<<gr((long long)H * D, TB), TB, 0, stream>>>(W1, WT16, D, H);
    {
        dim3 g(gr(Mpad, 256), H / 64);     // 256 M-rows per block, 64 N-cols
        k_wmma_gemm<<<g, TB, 0, stream>>>(A16, WT16, hA, Mpad, H, D);
    }
    // 3) aggregate + bias + relu -> hB
    k_self_init<<<gr((long long)N * (H / 4), TB), TB, 0, stream>>>(hA, hB, dinv, N, H);
    k_edge_agg <<<gr((long long)E * (H / 4), TB), TB, 0, stream>>>(hA, hB, src, dst, dinv, E, H);
    k_bias_act <<<gr((long long)N * H, TB), TB, 0, stream>>>(hB, b1, N, H, 1);

    // 4) layer 2 GEMM: hA = hB @ W2
    k_conv_pad<<<gr((long long)Mpad * H, TB), TB, 0, stream>>>(hB, A16, N, Mpad, H);
    k_conv_wt <<<gr((long long)H * H, TB), TB, 0, stream>>>(W2, WT16, H, H);
    {
        dim3 g(gr(Mpad, 256), H / 64);
        k_wmma_gemm<<<g, TB, 0, stream>>>(A16, WT16, hA, Mpad, H, H);
    }
    // 5) aggregate + bias (no relu) -> hB
    k_self_init<<<gr((long long)N * (H / 4), TB), TB, 0, stream>>>(hA, hB, dinv, N, H);
    k_edge_agg <<<gr((long long)E * (H / 4), TB), TB, 0, stream>>>(hA, hB, src, dst, dinv, E, H);
    k_bias_act <<<gr((long long)N * H, TB), TB, 0, stream>>>(hB, b2, N, H, 0);

    // 6) head: h3 = tanh(hB @ Wl + bl)
    k_head<<<gr((long long)N * A, TB), TB, 0, stream>>>(hB, Wl, bl, h3, N, H, A);

    // 7) global mean pool
    k_pool_zero<<<gr(NG * A, TB), TB, 0, stream>>>(out, cnt, NG * A, NG);
    k_pool     <<<gr(N, TB), TB, 0, stream>>>(h3, batch, out, cnt, N, A);
    k_pool_fin <<<gr(NG * A, TB), TB, 0, stream>>>(out, cnt, NG * A, A);
}
